// GRA_14955076125012
// MI455X (gfx1250) — matrix operand order repaired and checked by reading, compiled, not run on hardware
//
#include <hip/hip_runtime.h>
#include <hip/hip_bf16.h>
#include <stdint.h>

// ---------------------------------------------------------------------------
// CDNA5 (gfx1250) conv + criss-cross attention.
// GEMM work on v_wmma_f32_16x16x32_f16; k-panel staged to LDS via the
// Tensor Data Mover (tensor_load_to_lds + s_wait_tensorcnt).
// ---------------------------------------------------------------------------

typedef _Float16 f16;
typedef __attribute__((ext_vector_type(16))) _Float16 v16h;
typedef __attribute__((ext_vector_type(8)))  _Float16 v8h;
typedef __attribute__((ext_vector_type(8)))  float    v8f;
typedef uint32_t u32x4 __attribute__((ext_vector_type(4)));
typedef uint32_t u32x8 __attribute__((ext_vector_type(8)));

#define NEGC (-1.0e30f)

__device__ __forceinline__ v16h pack16(v8h lo, v8h hi) {
  v16h r;
#pragma unroll
  for (int i = 0; i < 8; ++i) { r[i] = lo[i]; r[8 + i] = hi[i]; }
  return r;
}

__device__ __forceinline__ v8f wmma_f16(v16h a, v16h b, v8f c) {
  return __builtin_amdgcn_wmma_f32_16x16x32_f16(false, a, false, b, (short)0, c,
                                                false, false);
}

// ---------------------------------------------------------------------------
// K0: repack weights to f16 GEMM layouts.
// ---------------------------------------------------------------------------
__global__ void k_pack(const float* __restrict__ Wconv, const float* __restrict__ Wq,
                       const float* __restrict__ Wk, const float* __restrict__ Wv,
                       f16* Wc16, f16* Wqk16, f16* Wv16) {
  const int total = 64 * 640 + 16 * 64 + 64 * 64;
  for (int e = blockIdx.x * blockDim.x + threadIdx.x; e < total;
       e += gridDim.x * blockDim.x) {
    if (e < 40960) {
      int m = e / 640, k = e % 640;
      float v = 0.f;
      if (k < 612) {
        int tap = k / 68, cin = k % 68;
        v = Wconv[(m * 68 + cin) * 9 + tap];
      }
      Wc16[e] = (f16)v;
    } else if (e < 40960 + 1024) {
      int e2 = e - 40960;
      int m = e2 / 64, c = e2 % 64;
      Wqk16[e2] = (f16)((m < 8) ? Wq[m * 64 + c] : Wk[(m - 8) * 64 + c]);
    } else {
      int e3 = e - 41984;
      Wv16[e3] = (f16)Wv[e3];
    }
  }
}

// ---------------------------------------------------------------------------
// K1: fused interleave + conv3x3(68->64)+bias+relu+residual -> x2, THEN the
// q/k/v projections directly off the freshly produced tile held in LDS.
// Implicit GEMM, im2col panel [32 px][640 K] in LDS; 8 waves = 4Mx2N tiles,
// 20 WMMA K-steps; projection: 10 wave-tasks (2 QK-fused + 8 V), K=64.
// ---------------------------------------------------------------------------
__global__ __launch_bounds__(256) void k_conv(
    const float* __restrict__ x, const float* __restrict__ y,
    const float* __restrict__ bconv, const f16* __restrict__ Wc16,
    const f16* __restrict__ Wqk16, const f16* __restrict__ Wv16,
    const float* __restrict__ bq, const float* __restrict__ bk,
    const float* __restrict__ bv, float* __restrict__ x2,
    f16* __restrict__ qh, f16* __restrict__ kh, f16* __restrict__ vh) {
  int bid = blockIdx.x;  // B * H * (W/32) = 8192
  int b = bid >> 9;
  int rem = bid & 511;
  int h = rem >> 2;
  int w0 = (rem & 3) * 32;

  __shared__ __align__(16) f16 Bpan[32 * 640];  // [n][k], 40KB
  __shared__ __align__(16) f16 Xpan[32 * 64];   // [n][c], 4KB (x2 tile, f16)

  int tid = threadIdx.x;
  for (int e = tid; e < 32 * 640; e += 256) {
    int n = e / 640, k = e % 640;
    float v = 0.f;
    if (k < 612) {
      int tap = k / 68, cin = k % 68;
      int hh = h + tap / 3 - 1, ww = w0 + n + tap % 3 - 1;
      if (hh >= 0 && hh < 128 && ww >= 0 && ww < 128) {
        int g = cin / 17, r = cin % 17;
        v = (r < 16) ? x[((b * 64 + g * 16 + r) * 128 + hh) * 128 + ww]
                     : y[(b * 128 + hh) * 128 + ww];
      }
    }
    Bpan[e] = (f16)v;
  }
  __syncthreads();

  int wave = tid >> 5, lane = tid & 31, laneHi = lane >> 4;
  int mtile = wave & 3, ntile = wave >> 2;
  int m = mtile * 16 + (lane & 15);
  int nn = ntile * 16 + (lane & 15);
  const f16* Wr = Wc16 + m * 640;
  const f16* Br = Bpan + nn * 640;

  v8f acc = {};
  for (int kc = 0; kc < 20; ++kc) {
    int ka = kc * 32 + laneHi * 8;
    v16h A = pack16(*(const v8h*)(Wr + ka), *(const v8h*)(Wr + ka + 16));
    int kB = kc * 32 + laneHi * 8;
    v16h Bf = pack16(*(const v8h*)(Br + kB), *(const v8h*)(Br + kB + 16));
    acc = wmma_f16(A, Bf, acc);
  }

  int wloc = w0 + nn;
#pragma unroll
  for (int r = 0; r < 8; ++r) {
    int ch = mtile * 16 + r + laneHi * 8;
    float o = acc[r] + bconv[ch];
    o = o > 0.f ? o : 0.f;
    int idx = ((b * 64 + ch) * 128 + h) * 128 + wloc;
    o += x[idx];
    x2[idx] = o;
    Xpan[nn * 64 + ch] = (f16)o;
  }
  __syncthreads();

  // ---- q/k/v projections from Xpan (K=64 -> 2 WMMA steps per task) ----
  for (int t = wave; t < 10; t += 8) {
    bool qk = (t < 2);
    int mt = qk ? 0 : (t - 2) & 3;
    int nt = qk ? t : (t - 2) >> 2;
    const f16* Arow = qk ? (Wqk16 + (lane & 15) * 64)
                         : (Wv16 + (mt * 16 + (lane & 15)) * 64);
    const f16* Br2 = Xpan + (nt * 16 + (lane & 15)) * 64;
    v8f a2 = {};
#pragma unroll
    for (int kc = 0; kc < 2; ++kc) {
      int ka = kc * 32 + laneHi * 8;
      v16h A = pack16(*(const v8h*)(Arow + ka), *(const v8h*)(Arow + ka + 16));
      int kB = kc * 32 + laneHi * 8;
      v16h Bf = pack16(*(const v8h*)(Br2 + kB), *(const v8h*)(Br2 + kB + 16));
      a2 = wmma_f16(A, Bf, a2);
    }
    int w = w0 + nt * 16 + (lane & 15);
#pragma unroll
    for (int r = 0; r < 8; ++r) {
      int mloc = r + laneHi * 8;
      if (qk) {
        if (mloc < 8)
          qh[((b * 8 + mloc) * 128 + h) * 128 + w] = (f16)(a2[r] + bq[mloc]);
        else
          kh[((b * 8 + mloc - 8) * 128 + h) * 128 + w] = (f16)(a2[r] + bk[mloc - 8]);
      } else {
        int ch = mt * 16 + mloc;
        vh[((b * 64 + ch) * 128 + h) * 128 + w] = (f16)(a2[r] + bv[ch]);
      }
    }
  }
}

// ---------------------------------------------------------------------------
// K2: LDS-tiled transpose  src[b][C][h][w] -> dst[b][w][C][h]
// (makes every later attention access contiguous; kills the 2B/256B gathers)
// ---------------------------------------------------------------------------
__global__ __launch_bounds__(256) void k_transpose(const f16* __restrict__ src,
                                                   f16* __restrict__ dst, int C) {
  int bid = blockIdx.x;  // b * C * 16 tiles of 32x32
  int b = bid / (C * 16);
  int rem = bid % (C * 16);
  int c = rem / 16;
  int tile = rem % 16;
  int h0 = (tile >> 2) * 32, w0 = (tile & 3) * 32;
  __shared__ f16 t[32][33];
  int i = threadIdx.x >> 5, j = threadIdx.x & 31;  // 8 x 32
  for (int ii = i; ii < 32; ii += 8)
    t[ii][j] = src[((b * C + c) * 128 + h0 + ii) * 128 + w0 + j];
  __syncthreads();
  for (int ii = i; ii < 32; ii += 8)
    dst[((b * 128 + w0 + ii) * C + c) * 128 + h0 + j] = t[j][ii];
}

// ---------------------------------------------------------------------------
// K3: scores + softmax. One block per (b, w). The whole k[b] (8x128 rows x
// 256B) is DMA'd into 278KB dynamic LDS by the Tensor Data Mover with
// pad_enable inserting 16B after every 256B row (bank-spread layout).
// Each of 128 threads owns a pixel; two-pass softmax writes UNNORMALIZED
// exp() f16 + per-pixel sum l (normalization folded into K6).
// ---------------------------------------------------------------------------
__global__ __launch_bounds__(256) void k_score(const f16* __restrict__ qT,
                                               const f16* __restrict__ kh,
                                               f16* __restrict__ att,
                                               float* __restrict__ lsum) {
  extern __shared__ __align__(16) char smem[];
  f16* kk = (f16*)smem;                            // [8][128][136]
  float* qc = (float*)(smem + 8 * 128 * 136 * 2);  // [8][128]

  int b = blockIdx.x >> 7, w = blockIdx.x & 127;
  int tid = threadIdx.x;
  int wave = tid >> 5;

  if (wave == 0) {
    // ---- Tensor Data Mover: k[b] (1024 rows x 128 f16) -> LDS, +16B pad/row
    uint32_t lds_off = (uint32_t)(uintptr_t)(void*)kk;
    uint64_t ga = (uint64_t)(uintptr_t)(kh + (size_t)b * 8 * 128 * 128);
    u32x4 g0;
    g0[0] = 1u;                                   // count=1 (valid user D#)
    g0[1] = lds_off;                              // lds_addr
    g0[2] = (uint32_t)ga;                         // global_addr[31:0]
    g0[3] = (uint32_t)(ga >> 32) | (2u << 30);    // addr[56:32] | type=2
    u32x8 g1;
    g1[0] = (1u << 16)        // data_size = 2B
          | (1u << 20)        // pad_enable
          | (5u << 22)        // pad_interval: 64 DWORDs (256B)
          | (3u << 25);       // pad_amount: 4 DWORDs (16B)
    g1[1] = (128u << 16);     // tensor_dim0 = 128 (bits 79:48, low half)
    g1[2] = (1024u << 16);    // tensor_dim1 = 1024 (bits 111:80, low half)
    g1[3] = (128u << 16);     // tile_dim0 = 128 (bits 127:112)
    g1[4] = 1024u;            // tile_dim1 = 1024, tile_dim2 = 0
    g1[5] = 128u;             // tensor_dim0_stride = 128
    g1[6] = 0u;
    g1[7] = 0u;
    asm volatile("tensor_load_to_lds %0, %1" ::"s"(g0), "s"(g1) : "memory");
    asm volatile("s_wait_tensorcnt 0x0" ::: "memory");
  }

  const f16* qsrc = qT + ((size_t)(b * 128 + w) * 8) * 128;  // [8][128] contig
  for (int e = tid; e < 1024; e += 256) qc[e] = (float)qsrc[e];
  __syncthreads();

  if (tid >= 128) return;
  int h = tid;
  float qp[8];
#pragma unroll
  for (int c = 0; c < 8; ++c) qp[c] = qc[c * 128 + h];

  // pass 1: max over 256 logits
  float mx = -3.0e38f;
  for (int j = 0; j < 128; ++j) {  // column (eH), broadcast LDS reads
    float ev = 0.f;
#pragma unroll
    for (int c = 0; c < 8; ++c) ev += qp[c] * (float)kk[(c * 128 + j) * 136 + w];
    if (j == h) ev += NEGC;
    mx = fmaxf(mx, ev);
  }
  for (int jb = 0; jb < 16; ++jb) {  // row (eW), vectorized
    v8h kv[8];
#pragma unroll
    for (int c = 0; c < 8; ++c)
      kv[c] = *(const v8h*)(kk + (c * 128 + h) * 136 + jb * 8);
#pragma unroll
    for (int i = 0; i < 8; ++i) {
      float ev = 0.f;
#pragma unroll
      for (int c = 0; c < 8; ++c) ev += qp[c] * (float)kv[c][i];
      mx = fmaxf(mx, ev);
    }
  }

  // pass 2: exp + sum; write unnormalized probabilities
  float l = 0.f;
  f16* op = att + (size_t)((b * 128 + h) * 128 + w) * 256;
  for (int j = 0; j < 128; ++j) {
    float ev = 0.f;
#pragma unroll
    for (int c = 0; c < 8; ++c) ev += qp[c] * (float)kk[(c * 128 + j) * 136 + w];
    if (j == h) ev += NEGC;
    float p = __expf(ev - mx);
    l += p;
    op[j] = (f16)p;
  }
  for (int jb = 0; jb < 16; ++jb) {
    v8h kv[8];
#pragma unroll
    for (int c = 0; c < 8; ++c)
      kv[c] = *(const v8h*)(kk + (c * 128 + h) * 136 + jb * 8);
#pragma unroll
    for (int i = 0; i < 8; ++i) {
      float ev = 0.f;
#pragma unroll
      for (int c = 0; c < 8; ++c) ev += qp[c] * (float)kv[c][i];
      float p = __expf(ev - mx);
      l += p;
      op[128 + jb * 8 + i] = (f16)p;
    }
  }
  lsum[(b * 128 + h) * 128 + w] = l;
}

// ---------------------------------------------------------------------------
// K4: column aggregation outH[c,h] = sum_j attH[h,j]*v[c,j,w] per (b,w).
// A operand from vT (contiguous), B operand = contiguous att rows. Pure
// global->WMMA streaming, no LDS staging needed.
// ---------------------------------------------------------------------------
__global__ __launch_bounds__(256) void k_aggH(const f16* __restrict__ vT,
                                              const f16* __restrict__ att,
                                              float* __restrict__ accb) {
  int b = blockIdx.x >> 7, w = blockIdx.x & 127;
  int tid = threadIdx.x;
  int wave = tid >> 5, lane = tid & 31, laneHi = lane >> 4;
  int mtile = wave & 3;
  int m = mtile * 16 + (lane & 15);
  const f16* Ar = vT + ((size_t)(b * 128 + w) * 64 + m) * 128;  // v[c=m][j]

  for (int nt = wave >> 2; nt < 8; nt += 2) {
    int hh = nt * 16 + (lane & 15);
    const f16* bb = att + (size_t)((b * 128 + hh) * 128 + w) * 256;
    __builtin_prefetch(bb, 0, 3);
    v8f acc = {};
#pragma unroll
    for (int kc = 0; kc < 4; ++kc) {
      int ka = kc * 32 + laneHi * 8;
      v16h A = pack16(*(const v8h*)(Ar + ka), *(const v8h*)(Ar + ka + 16));
      int kB = kc * 32 + laneHi * 8;
      v16h Bf = pack16(*(const v8h*)(bb + kB), *(const v8h*)(bb + kB + 16));
      acc = wmma_f16(A, Bf, acc);
    }
#pragma unroll
    for (int r = 0; r < 8; ++r) {
      int ch = mtile * 16 + r + laneHi * 8;
      accb[((b * 64 + ch) * 128 + hh) * 128 + w] = acc[r];
    }
  }
}

// ---------------------------------------------------------------------------
// K5: row aggregation outW[c,w] = sum_j attW[w,j]*v[c,h,j] per (b,h).
// Both operands contiguous in global memory. Accumulates into accb.
// ---------------------------------------------------------------------------
__global__ __launch_bounds__(256) void k_aggW(const f16* __restrict__ vh,
                                              const f16* __restrict__ att,
                                              float* __restrict__ accb) {
  int b = blockIdx.x >> 7, h = blockIdx.x & 127;
  int tid = threadIdx.x;
  int wave = tid >> 5, lane = tid & 31, laneHi = lane >> 4;
  int mtile = wave & 3;
  int m = mtile * 16 + (lane & 15);
  const f16* Ar = vh + ((b * 64 + m) * 128 + h) * 128;  // v[c=m][h][j]

  for (int nt = wave >> 2; nt < 8; nt += 2) {
    int ww = nt * 16 + (lane & 15);
    const f16* bb = att + (size_t)((b * 128 + h) * 128 + ww) * 256 + 128;
    __builtin_prefetch(bb, 0, 3);
    v8f acc = {};
#pragma unroll
    for (int kc = 0; kc < 4; ++kc) {
      int ka = kc * 32 + laneHi * 8;
      v16h A = pack16(*(const v8h*)(Ar + ka), *(const v8h*)(Ar + ka + 16));
      int kB = kc * 32 + laneHi * 8;
      v16h Bf = pack16(*(const v8h*)(bb + kB), *(const v8h*)(bb + kB + 16));
      acc = wmma_f16(A, Bf, acc);
    }
#pragma unroll
    for (int r = 0; r < 8; ++r) {
      int ch = mtile * 16 + r + laneHi * 8;
      int idx = ((b * 64 + ch) * 128 + h) * 128 + ww;
      accb[idx] += acc[r];
    }
  }
}

// ---------------------------------------------------------------------------
// K6: x3 = gamma * (accb / l) + x2   (softmax normalization folded in here)
// ---------------------------------------------------------------------------
__global__ void k_combine(const float* __restrict__ x2, const float* __restrict__ accb,
                          const float* __restrict__ lsum, const float* __restrict__ gamma,
                          float* __restrict__ out_x3) {
  int idx = blockIdx.x * 256 + threadIdx.x;
  if (idx >= 16777216) return;
  int w = idx & 127;
  int h = (idx >> 7) & 127;
  int b = idx >> 20;
  float l = lsum[(b * 128 + h) * 128 + w];
  out_x3[idx] = gamma[0] * (accb[idx] / l) + x2[idx];
}

// ---------------------------------------------------------------------------
// K7: y2 = y + conv3x3(x3, Wscore) + bscore   (64->1, tiny, VALU)
// ---------------------------------------------------------------------------
__global__ void k_score_conv(const float* __restrict__ x3, const float* __restrict__ y,
                             const float* __restrict__ Wscore,
                             const float* __restrict__ bscore, float* __restrict__ y2) {
  int idx = blockIdx.x * 256 + threadIdx.x;
  if (idx >= 262144) return;
  int b = idx >> 14;
  int h = (idx >> 7) & 127;
  int w = idx & 127;
  float acc = bscore[0];
  for (int c = 0; c < 64; ++c) {
    const float* xp = x3 + ((b * 64 + c) * 128) * 128;
    const float* wp = Wscore + c * 9;
#pragma unroll
    for (int ky = 0; ky < 3; ++ky) {
      int hh = h + ky - 1;
      if (hh < 0 || hh >= 128) continue;
#pragma unroll
      for (int kx = 0; kx < 3; ++kx) {
        int ww = w + kx - 1;
        if (ww < 0 || ww >= 128) continue;
        acc += xp[hh * 128 + ww] * wp[ky * 3 + kx];
      }
    }
  }
  y2[idx] = y[idx] + acc;
}

// ---------------------------------------------------------------------------
extern "C" void kernel_launch(void* const* d_in, const int* in_sizes, int n_in,
                              void* d_out, int out_size, void* d_ws, size_t ws_size,
                              hipStream_t stream) {
  (void)in_sizes; (void)n_in; (void)out_size; (void)ws_size;
  const float* x      = (const float*)d_in[0];
  const float* y      = (const float*)d_in[1];
  const float* Wconv  = (const float*)d_in[2];
  const float* bconv  = (const float*)d_in[3];
  const float* Wscore = (const float*)d_in[4];
  const float* bscore = (const float*)d_in[5];
  const float* Wq     = (const float*)d_in[6];
  const float* bq     = (const float*)d_in[7];
  const float* Wk     = (const float*)d_in[8];
  const float* bk     = (const float*)d_in[9];
  const float* Wv     = (const float*)d_in[10];
  const float* bv     = (const float*)d_in[11];
  const float* gamma  = (const float*)d_in[12];

  uint8_t* ws = (uint8_t*)d_ws;
  f16*   Wc16  = (f16*)(ws + 0);            //      81,920 B
  f16*   Wqk16 = (f16*)(ws + 81920);        //       2,048 B
  f16*   Wv16  = (f16*)(ws + 83968);        //       8,192 B
  float* x2    = (float*)(ws + 92160);      //  67,108,864 B
  f16*   qh    = (f16*)(ws + 67201024);     //   4,194,304 B
  f16*   kh    = (f16*)(ws + 71395328);     //   4,194,304 B
  f16*   vh    = (f16*)(ws + 75589632);     //  33,554,432 B
  f16*   vT    = (f16*)(ws + 109144064);    //  33,554,432 B  [b][w][c][h]
  f16*   qT    = (f16*)(ws + 142698496);    //   4,194,304 B  [b][w][c][h]
  f16*   att   = (f16*)(ws + 146892800);    // 134,217,728 B
  float* accb  = (float*)(ws + 281110528);  //  67,108,864 B
  float* lsum  = (float*)(ws + 348219392);  //   1,048,576 B  (total ~349 MB)

  float* out_x3 = (float*)d_out;
  float* out_y2 = (float*)d_out + 16777216;

  k_pack<<<64, 256, 0, stream>>>(Wconv, Wq, Wk, Wv, Wc16, Wqk16, Wv16);
  k_conv<<<8192, 256, 0, stream>>>(x, y, bconv, Wc16, Wqk16, Wv16, bq, bk, bv,
                                   x2, qh, kh, vh);
  k_transpose<<<16384, 256, 0, stream>>>(vh, vT, 64);
  k_transpose<<<2048, 256, 0, stream>>>(qh, qT, 8);
  k_score<<<2048, 256, 282624, stream>>>(qT, kh, att, lsum);  // 278KB TDM + 4KB q
  k_aggH<<<2048, 256, 0, stream>>>(vT, att, accb);
  k_aggW<<<2048, 256, 0, stream>>>(vh, att, accb);
  k_combine<<<65536, 256, 0, stream>>>(x2, accb, lsum, gamma, out_x3);
  k_score_conv<<<1024, 256, 0, stream>>>(out_x3, y, Wscore, bscore, out_y2);
}
